// Recommender_84954453115299
// MI455X (gfx1250) — matrix operand order repaired and checked by reading, not compiled, not run on hardware
//
#include <hip/hip_runtime.h>
#include <math.h>

#define N_USERS   200000
#define N_ITEMS   300000
#define N_NODES   600000
#define N_EDGES   10000000
#define DIM       64
#define BATCH     4096
#define K_NEG     4
#define DECAY     1.0e-4f

typedef __attribute__((ext_vector_type(16))) _Float16 v16h;
typedef __attribute__((ext_vector_type(8)))  float    v8f;
typedef __attribute__((ext_vector_type(4)))  float    f4;   // native vector for NT builtins

// ---------------------------------------------------------------------------
// Init: acc = all_embed (NT stores: acc is streamed, keep it out of L2 so the
// gather/atomic feature buffers stay resident), xA = xB = 0 (regular stores:
// these are the hot atomic targets), scalar accumulators = 0.
// Re-runs every launch so graph replay is deterministic.
// ---------------------------------------------------------------------------
__global__ void gcn_init_kernel(const float* __restrict__ emb,
                                float* __restrict__ acc,
                                float* __restrict__ xA,
                                float* __restrict__ xB,
                                float* __restrict__ scal) {
    size_t tid    = (size_t)blockIdx.x * blockDim.x + threadIdx.x;
    size_t stride = (size_t)gridDim.x * blockDim.x;
    if (tid < 16) scal[tid] = 0.0f;
    const size_t n4 = (size_t)N_NODES * DIM / 4;
    const f4* e4 = (const f4*)emb;
    f4* a4 = (f4*)acc;
    f4* A4 = (f4*)xA;
    f4* B4 = (f4*)xB;
    f4 z = {0.f, 0.f, 0.f, 0.f};
    for (size_t j = tid; j < n4; j += stride) {
        f4 e = __builtin_nontemporal_load(e4 + j);
        __builtin_nontemporal_store(e, a4 + j);
        A4[j] = z;
        B4[j] = z;
    }
}

// ---------------------------------------------------------------------------
// SpMM hop. Each wave grabs 32 consecutive edges with coalesced NT loads
// (edge arrays are streamed once -> TH=NT keeps the 153.6MB feature matrix
// resident in the 192MB L2), then broadcasts each edge's (row,col,val) to
// SGPRs with v_readlane (constant lane index after unrolling) -> scalar base
// addressing for the gather and atomic destination, no LDS-pipe traffic.
// All 32 lanes then gather one 64-dim row (float2/lane, 256B coalesced) and
// scatter-add with two global_atomic_add_f32 per lane.
// N_EDGES % 32 == 0 -> no tail, EXEC stays all-1s.
// ---------------------------------------------------------------------------
__global__ void spmm_kernel(const int* __restrict__ erow,
                            const int* __restrict__ ecol,
                            const float* __restrict__ eval_,
                            const float* __restrict__ x,
                            float* __restrict__ y) {
    const int lane = threadIdx.x & 31;
    size_t wid = ((size_t)blockIdx.x * blockDim.x + threadIdx.x) >> 5;
    size_t nw  = ((size_t)gridDim.x * blockDim.x) >> 5;
    for (size_t base = wid * 32; base < (size_t)N_EDGES; base += nw * 32) {
        int   r  = __builtin_nontemporal_load(erow + base + lane);
        int   c  = __builtin_nontemporal_load(ecol + base + lane);
        int   vb = __builtin_nontemporal_load((const int*)eval_ + base + lane);
#pragma unroll
        for (int j = 0; j < 32; ++j) {
            int   rj = __builtin_amdgcn_readlane(r, j);
            int   cj = __builtin_amdgcn_readlane(c, j);
            float vj = __int_as_float(__builtin_amdgcn_readlane(vb, j));
            const float2* src = (const float2*)(x + (size_t)cj * DIM);
            float2 m = src[lane];
            float* dst = y + (size_t)rj * DIM + lane * 2;
            atomicAdd(dst + 0, vj * m.x);
            atomicAdd(dst + 1, vj * m.y);
        }
    }
}

// ---------------------------------------------------------------------------
// acc += addsrc; optionally zero a buffer for the next hop's scatter target.
// acc is streamed (NT read/write); addsrc stays regular (it is the next hop's
// gather source and should remain L2-resident).
// ---------------------------------------------------------------------------
__global__ void update_kernel(float* __restrict__ acc,
                              const float* __restrict__ addsrc,
                              float* __restrict__ zerobuf) {
    size_t tid    = (size_t)blockIdx.x * blockDim.x + threadIdx.x;
    size_t stride = (size_t)gridDim.x * blockDim.x;
    const size_t n4 = (size_t)N_NODES * DIM / 4;
    f4* a4 = (f4*)acc;
    const f4* s4 = (const f4*)addsrc;
    f4* z4 = (f4*)zerobuf;
    f4 z = {0.f, 0.f, 0.f, 0.f};
    const bool do_zero = (zerobuf != nullptr);
    for (size_t j = tid; j < n4; j += stride) {
        f4 a = __builtin_nontemporal_load(a4 + j);
        f4 s = s4[j];
        a = a + s;
        __builtin_nontemporal_store(a, a4 + j);
        if (do_zero) z4[j] = z;
    }
}

// ---------------------------------------------------------------------------
// Scoring with WMMA. One wave handles a tile of 16 batch rows.
//   S_pos = U(16x64) x Pos^T(64x16),  S_neg = U x Navg^T
// where Navg = mean over K_NEG negatives (dot is linear, so
// mean_k dot(u,n_k) == dot(u, mean_k n_k)). K=64 -> two chained
// v_wmma_f32_16x16x32_f16. The diagonal of each 16x16 tile gives the
// per-row scores; lanes 0-7 hold diag M=0..7 (VGPR=lane), lanes 24-31
// hold diag M=8..15 (VGPR=lane-24), per the CDNA5 C/D layout.
// mf contribution per row: softplus(-(pos - negmean)).
// ---------------------------------------------------------------------------
__global__ void score_kernel(const float* __restrict__ acc,
                             const int* __restrict__ user,
                             const int* __restrict__ pos,
                             const int* __restrict__ neg,
                             float* __restrict__ mf_acc) {
    const int lane = threadIdx.x & 31;
    const int tile = blockIdx.x * (blockDim.x >> 5) + (threadIdx.x >> 5);
    const int MN   = lane & 15;            // row for A, column for B

    const float* ua = acc + (size_t)user[tile * 16 + MN] * DIM;
    const float* pa = acc + (size_t)(N_USERS + pos[tile * 16 + MN]) * DIM;
    const int*   nr = neg + (size_t)(tile * 16 + MN) * K_NEG;
    const float* n0 = acc + (size_t)(N_USERS + nr[0]) * DIM;
    const float* n1 = acc + (size_t)(N_USERS + nr[1]) * DIM;
    const float* n2 = acc + (size_t)(N_USERS + nr[2]) * DIM;
    const float* n3 = acc + (size_t)(N_USERS + nr[3]) * DIM;

    const int ahi = (lane >= 16) ? 8 : 0;   // A: half-wave K sub-offset
    const int bko = (lane >= 16) ? 16 : 0;  // B: twice the half-wave K offset (halved in use)

    v8f cp = {};
    v8f cn = {};
#pragma unroll
    for (int kb = 0; kb < DIM; kb += 32) {
        v16h a, bp, bn;
#pragma unroll
        for (int j = 0; j < 16; ++j) {
            // 16-bit A 16x32 layout: elems 0..7 -> K=(0..7)+ahi+kb,
            //                        elems 8..15 -> K=(16..23)+ahi+kb
            int k = kb + ahi + ((j < 8) ? j : (j + 8));
            a[j] = (_Float16)ua[k];
        }
#pragma unroll
        for (int j = 0; j < 16; ++j) {
            int k = kb + (bko >> 1) + ((j < 8) ? j : (j + 8));
            bp[j] = (_Float16)pa[k];
            bn[j] = (_Float16)(0.25f * (n0[k] + n1[k] + n2[k] + n3[k]));
        }
        cp = __builtin_amdgcn_wmma_f32_16x16x32_f16(false, a, false, bp,
                                                    (short)0, cp, false, false);
        cn = __builtin_amdgcn_wmma_f32_16x16x32_f16(false, a, false, bn,
                                                    (short)0, cn, false, false);
    }

    float dp[8], dn[8];
#pragma unroll
    for (int i = 0; i < 8; ++i) { dp[i] = cp[i]; dn[i] = cn[i]; }

    int di = (lane < 8) ? lane : ((lane >= 24) ? (lane - 24) : -1);
    if (di >= 0) {
        float diff = dp[di] - dn[di];
        float t  = -diff;
        float sp = fmaxf(t, 0.0f) + log1pf(__expf(-fabsf(t)));  // softplus(-diff)
        atomicAdd(mf_acc, sp);
    }
}

// ---------------------------------------------------------------------------
// Regularizer: sum of squares of the ORIGINAL embeddings at the sampled rows.
// One wave per row (24576 rows), float2 per lane, wave-reduce, 1 atomic/wave.
// ---------------------------------------------------------------------------
__global__ void reg_kernel(const float* __restrict__ emb,
                           const int* __restrict__ user,
                           const int* __restrict__ pos,
                           const int* __restrict__ neg,
                           float* __restrict__ reg_acc) {
    const int lane = threadIdx.x & 31;
    int wid = blockIdx.x * (blockDim.x >> 5) + (threadIdx.x >> 5);
    int nw  = gridDim.x * (blockDim.x >> 5);
    const int TOT = BATCH + BATCH + BATCH * K_NEG;   // 24576 rows
    float s = 0.0f;
    for (int r = wid; r < TOT; r += nw) {
        int node;
        if (r < BATCH)            node = user[r];
        else if (r < 2 * BATCH)   node = N_USERS + pos[r - BATCH];
        else                      node = N_USERS + neg[r - 2 * BATCH];
        const float2* src = (const float2*)(emb + (size_t)node * DIM);
        float2 v = src[lane];
        s += v.x * v.x + v.y * v.y;
    }
#pragma unroll
    for (int off = 16; off > 0; off >>= 1) s += __shfl_down(s, off, 32);
    if (lane == 0) atomicAdd(reg_acc, s);
}

// ---------------------------------------------------------------------------
// Finalize: 3 output scalars.
// ---------------------------------------------------------------------------
__global__ void finalize_kernel(const float* __restrict__ scal,
                                float* __restrict__ out) {
    if (blockIdx.x == 0 && threadIdx.x == 0) {
        float mf  = scal[0] / (float)BATCH;
        float emb = DECAY * (scal[1] * 0.5f) / (float)BATCH;
        out[0] = mf + emb;
        out[1] = mf;
        out[2] = emb;
    }
}

extern "C" void kernel_launch(void* const* d_in, const int* in_sizes, int n_in,
                              void* d_out, int out_size, void* d_ws, size_t ws_size,
                              hipStream_t stream) {
    (void)in_sizes; (void)n_in; (void)out_size; (void)ws_size;
    const float* all_embed = (const float*)d_in[0];
    const float* edge_val  = (const float*)d_in[1];
    const int*   edge_row  = (const int*)d_in[2];
    const int*   edge_col  = (const int*)d_in[3];
    const int*   user      = (const int*)d_in[4];
    const int*   pos_item  = (const int*)d_in[5];
    const int*   neg_item  = (const int*)d_in[6];
    float* out = (float*)d_out;

    const size_t NELEM = (size_t)N_NODES * DIM;
    char*  ws   = (char*)d_ws;
    float* scal = (float*)ws;                 // [0]=mf sum, [1]=reg sum
    float* acc  = (float*)(ws + 256);
    float* xA   = acc + NELEM;
    float* xB   = xA + NELEM;

    gcn_init_kernel<<<2048, 256, 0, stream>>>(all_embed, acc, xA, xB, scal);

    // hop 0: gather from original embeddings
    spmm_kernel<<<2048, 256, 0, stream>>>(edge_row, edge_col, edge_val, all_embed, xA);
    update_kernel<<<2048, 256, 0, stream>>>(acc, xA, nullptr);
    // hop 1
    spmm_kernel<<<2048, 256, 0, stream>>>(edge_row, edge_col, edge_val, xA, xB);
    update_kernel<<<2048, 256, 0, stream>>>(acc, xB, xA);   // also zero xA for hop 2
    // hop 2
    spmm_kernel<<<2048, 256, 0, stream>>>(edge_row, edge_col, edge_val, xB, xA);
    update_kernel<<<2048, 256, 0, stream>>>(acc, xA, nullptr);

    // BATCH=4096 rows -> 256 tiles of 16; 32 blocks x 8 waves = 256 waves
    score_kernel<<<32, 256, 0, stream>>>(acc, user, pos_item, neg_item, scal + 0);
    reg_kernel<<<64, 256, 0, stream>>>(all_embed, user, pos_item, neg_item, scal + 1);
    finalize_kernel<<<1, 64, 0, stream>>>(scal, out);
}
